// Model_39676907881591
// MI455X (gfx1250) — matrix operand order repaired and checked by reading, not compiled, not run on hardware
//
#include <hip/hip_runtime.h>
#include <stdint.h>

#define BHN 32
#define SEQ 2048
#define DH  128

typedef __attribute__((ext_vector_type(16))) __bf16 v16bf;
typedef __attribute__((ext_vector_type(8)))  float  v8f;

union VB { uint32_t u[8]; v16bf v; };

__device__ __forceinline__ uint32_t pk_bf16(float lo, float hi) {
  union { __bf16 h[2]; uint32_t u; } t;
  t.h[0] = (__bf16)lo;
  t.h[1] = (__bf16)hi;
  return t.u;
}

__launch_bounds__(256, 1)
__global__ void fa_fwd_kernel(const float* __restrict__ q,
                              const float* __restrict__ k,
                              const float* __restrict__ v,
                              const float* __restrict__ sf,
                              const int*   __restrict__ dpp,
                              float* __restrict__ out) {
  // K chunk: 32 rows x 128 d as bf16 pairs, 65 uints/row (260B stride -> +1 bank/row)
  __shared__ uint32_t ksu[32 * 65];
  // V^T chunk: 128 d-rows x 32 k, bf16, stride 34 (68B -> +17 banks/row)
  __shared__ __attribute__((aligned(16))) __bf16 vsb[128 * 34];
  // Per-wave P tile: 16 q-rows x 32 k, bf16, stride 34
  __shared__ __attribute__((aligned(16))) __bf16 psb[8 * 16 * 34];

  const int tid  = threadIdx.x;
  const int lane = tid & 31;
  const int wv   = tid >> 5;
  const int half = lane >> 4;   // which 16-lane half
  const int n    = lane & 15;   // M for A-operand, N for B/C/D operands

  const int bh    = blockIdx.y;
  const int qbase = blockIdx.x * 128 + wv * 16;

  // ---- Load 16x128 Q strip into bf16 A-operand layout (4 chunks of K=32) ----
  const float* Qrow = q + ((size_t)bh * SEQ + (qbase + n)) * DH;
  VB qa[4];
#pragma unroll
  for (int c = 0; c < 4; ++c) {
#pragma unroll
    for (int j = 0; j < 8; ++j) {
      const int k0 = half * 16 + 2 * j;
      const int d  = 32 * c + k0;
      qa[c].u[j] = pk_bf16(Qrow[d], Qrow[d + 1]);
    }
  }

  v8f oacc[8];
#pragma unroll
  for (int dt = 0; dt < 8; ++dt) oacc[dt] = (v8f){0.f,0.f,0.f,0.f,0.f,0.f,0.f,0.f};
  float mrow[8], lrow[8];
#pragma unroll
  for (int r = 0; r < 8; ++r) { mrow[r] = -3.0e38f; lrow[r] = 0.0f; }

  // Per-thread slice of the cooperative 32x128 chunk fill:
  // 1024 float4's per chunk; this thread owns t4 = tid + 256*f, f=0..3.
  const int row_f[4] = { (tid + 0) >> 5, (tid + 256) >> 5, (tid + 512) >> 5, (tid + 768) >> 5 };
  const int d4_f     = (tid & 31) << 2;   // same d-offset for all 4 slices

  const float* kbase_p = k + (size_t)bh * SEQ * DH;
  const float* vbase_p = v + (size_t)bh * SEQ * DH;

  // ---- prologue: load chunk 0 into registers (double buffer) ----
  float4 kreg[4], vreg[4];
#pragma unroll
  for (int f = 0; f < 4; ++f) {
    kreg[f] = *(const float4*)(kbase_p + row_f[f] * DH + d4_f);
    vreg[f] = *(const float4*)(vbase_p + row_f[f] * DH + d4_f);
  }

  for (int kb = 0; kb < SEQ; kb += 32) {
    // ---- drain register buffer into LDS (bf16 convert; K row-major, V transposed) ----
#pragma unroll
    for (int f = 0; f < 4; ++f) {
      const int row = row_f[f];
      ksu[row * 65 + (d4_f >> 1) + 0] = pk_bf16(kreg[f].x, kreg[f].y);
      ksu[row * 65 + (d4_f >> 1) + 1] = pk_bf16(kreg[f].z, kreg[f].w);
      vsb[(d4_f + 0) * 34 + row] = (__bf16)vreg[f].x;
      vsb[(d4_f + 1) * 34 + row] = (__bf16)vreg[f].y;
      vsb[(d4_f + 2) * 34 + row] = (__bf16)vreg[f].z;
      vsb[(d4_f + 3) * 34 + row] = (__bf16)vreg[f].w;
    }
    __syncthreads();   // chunk visible to all waves

    // ---- issue global loads for the NEXT chunk now; latency overlaps compute ----
    if (kb + 32 < SEQ) {
      const float* kc = kbase_p + (size_t)(kb + 32) * DH;
      const float* vc = vbase_p + (size_t)(kb + 32) * DH;
#pragma unroll
      for (int f = 0; f < 4; ++f) {
        kreg[f] = *(const float4*)(kc + row_f[f] * DH + d4_f);
        vreg[f] = *(const float4*)(vc + row_f[f] * DH + d4_f);
      }
    }

    // ---- S = Q @ K^T for this 16x32 strip (2 output tiles, 4 D-chunks) ----
    v8f s0 = (v8f){0.f,0.f,0.f,0.f,0.f,0.f,0.f,0.f};
    v8f s1 = (v8f){0.f,0.f,0.f,0.f,0.f,0.f,0.f,0.f};
#pragma unroll
    for (int c = 0; c < 4; ++c) {
      VB b0, b1;
#pragma unroll
      for (int j = 0; j < 8; ++j) {
        const int off = 16 * c + 8 * half + j;   // pair index within K row
        b0.u[j] = ksu[n * 65 + off];             // K rows kb+n
        b1.u[j] = ksu[(16 + n) * 65 + off];      // K rows kb+16+n
      }
      s0 = __builtin_amdgcn_wmma_f32_16x16x32_bf16(false, qa[c].v, false, b0.v,
                                                   (short)0, s0, false, false);
      s1 = __builtin_amdgcn_wmma_f32_16x16x32_bf16(false, qa[c].v, false, b1.v,
                                                   (short)0, s1, false, false);
    }

    // ---- elementwise scale + online softmax (fp32) ----
    float p0[8], p1[8], corr[8];
#pragma unroll
    for (int r = 0; r < 8; ++r) {
      const size_t srow = (size_t)(qbase + r + 8 * half) * SEQ + kb;
      const float a = s0[r] * sf[srow + n];
      const float b = s1[r] * sf[srow + 16 + n];
      float mx = fmaxf(a, b);
      mx = fmaxf(mx, __shfl_xor(mx, 1, 32));
      mx = fmaxf(mx, __shfl_xor(mx, 2, 32));
      mx = fmaxf(mx, __shfl_xor(mx, 4, 32));
      mx = fmaxf(mx, __shfl_xor(mx, 8, 32));
      const float mnew = fmaxf(mrow[r], mx);
      const float c0   = __expf(mrow[r] - mnew);
      const float e0   = __expf(a - mnew);
      const float e1   = __expf(b - mnew);
      float rs = e0 + e1;
      rs += __shfl_xor(rs, 1, 32);
      rs += __shfl_xor(rs, 2, 32);
      rs += __shfl_xor(rs, 4, 32);
      rs += __shfl_xor(rs, 8, 32);
      lrow[r] = lrow[r] * c0 + rs;
      mrow[r] = mnew;
      corr[r] = c0; p0[r] = e0; p1[r] = e1;
    }
#pragma unroll
    for (int dt = 0; dt < 8; ++dt)
#pragma unroll
      for (int r = 0; r < 8; ++r)
        oacc[dt][r] *= corr[r];

    // ---- P (C-layout) -> wave-private LDS -> A-operand layout ----
    __bf16* pw = psb + wv * (16 * 34);
#pragma unroll
    for (int r = 0; r < 8; ++r) {
      pw[(r + 8 * half) * 34 + n]      = (__bf16)p0[r];
      pw[(r + 8 * half) * 34 + 16 + n] = (__bf16)p1[r];
    }
    asm volatile("s_wait_dscnt 0" ::: "memory");  // within-wave LDS RAW
    VB pa;
#pragma unroll
    for (int j = 0; j < 8; ++j) {
      const int k0 = half * 16 + 2 * j;
      pa.u[j] = *(const uint32_t*)&pw[n * 34 + k0];
    }

    // ---- O += P @ V (8 d-tiles, K=32) ----
#pragma unroll
    for (int dt = 0; dt < 8; ++dt) {
      VB bv;
#pragma unroll
      for (int j = 0; j < 8; ++j)
        bv.u[j] = *(const uint32_t*)&vsb[(16 * dt + n) * 34 + 16 * half + 2 * j];
      oacc[dt] = __builtin_amdgcn_wmma_f32_16x16x32_bf16(false, pa.v, false, bv.v,
                                                         (short)0, oacc[dt], false, false);
    }

    __syncthreads();   // all waves done reading LDS before next fill overwrites it
  }

  // ---- epilogue: out = O / l * dropout_p, fp32 coalesced stores ----
  const float dp = (float)(*dpp);
#pragma unroll
  for (int r = 0; r < 8; ++r) {
    const float inv = dp / lrow[r];
    const size_t orow = ((size_t)bh * SEQ + (qbase + r + 8 * half)) * DH;
#pragma unroll
    for (int dt = 0; dt < 8; ++dt)
      out[orow + 16 * dt + n] = oacc[dt][r] * inv;
  }
}

extern "C" void kernel_launch(void* const* d_in, const int* in_sizes, int n_in,
                              void* d_out, int out_size, void* d_ws, size_t ws_size,
                              hipStream_t stream) {
  (void)in_sizes; (void)n_in; (void)d_ws; (void)ws_size; (void)out_size;
  const float* q  = (const float*)d_in[0];
  const float* k  = (const float*)d_in[1];
  const float* v  = (const float*)d_in[2];
  const float* sf = (const float*)d_in[3];
  const int*   dp = (const int*)d_in[4];
  float* out = (float*)d_out;
  dim3 grid(SEQ / 128, BHN);
  dim3 block(256);
  fa_fwd_kernel<<<grid, block, 0, stream>>>(q, k, v, sf, dp, out);
}
